// PromptEncoder_9947144258105
// MI455X (gfx1250) — matrix operand order repaired and checked by reading, not compiled, not run on hardware
//
#include <hip/hip_runtime.h>
#include <hip/hip_bf16.h>
#include <stdint.h>

typedef __attribute__((ext_vector_type(16))) __bf16 v16bf;
typedef __attribute__((ext_vector_type(8)))  float  v8f;
typedef __attribute__((ext_vector_type(4)))  unsigned int v4u;
typedef __attribute__((ext_vector_type(8)))  int  v8i;
typedef __attribute__((ext_vector_type(4)))  int  v4i;

#define HDIM 768
#define MROWS 32           // rows per block (2 WMMA M-tiles)
#define MTILES 2
#define KT   24            // K tiles of 32  (768/32)
#define NT   48            // N tiles of 16  (768/16)
#define NTW  6             // N tiles per wave (48 / 8 waves)
#define THREADS 256
#define EPS 1e-5f

// WGP-scope prefetch: pulls into all cache levels (builtin only gives SE/SYS).
__device__ __forceinline__ void prefetch_wgp(const void* p) {
  asm volatile("global_prefetch_b8 %0, off" :: "v"(p) : "memory");
}

// ---------------------------------------------------------------------------
// Prep: W[o,h] f32 (row-major) -> fragment-swizzled bf16 hi/lo.
// B-fragment layout for v_wmma_f32_16x16x32_bf16 (32x16 B, bf16):
//   lane L (0..15):  column n = n0 + L,  K = k0 + {0..7, 16..23}  (16 bf16, 32 bytes)
//   lane L+16:       column n = n0 + L,  K = k0 + {8..15, 24..31}
// Index: frag[tile*512 + lane*16 + j], tile = ntile*KT + ktile.
// ---------------------------------------------------------------------------
__global__ void prep_w_kernel(const float* __restrict__ W,
                              __bf16* __restrict__ whi,
                              __bf16* __restrict__ wlo) {
  int idx  = blockIdx.x * blockDim.x + threadIdx.x;   // 0 .. 768*768-1
  int tile = idx >> 9;
  int r    = idx & 511;
  int lane = r >> 4;
  int j    = r & 15;
  int nt   = tile / KT;
  int kt   = tile % KT;
  int o    = nt * 16 + (lane & 15);
  int k    = kt * 32 + (lane >> 4) * 8 + (j & 7) + (j >> 3) * 16;
  float w  = W[o * HDIM + k];
  __bf16 hi = (__bf16)w;
  whi[idx] = hi;
  wlo[idx] = (__bf16)(w - (float)hi);
}

// ---------------------------------------------------------------------------
// Fused: z = x + x*W^T + b ; y = LayerNorm(z)*gamma+beta ; out = mask? y : x
// ---------------------------------------------------------------------------
__global__ void __launch_bounds__(THREADS)
fused_gemm_ln_kernel(const float* __restrict__ x,
                     const int*   __restrict__ mask,
                     const __bf16* __restrict__ whi,
                     const __bf16* __restrict__ wlo,
                     const float* __restrict__ bias,
                     const float* __restrict__ gamma,
                     const float* __restrict__ beta,
                     float* __restrict__ out) {
  // 96 KB: A fragments (bf16 hi/lo, 2 M-tiles) aliased later by z (f32)
  __shared__ __align__(32) unsigned char smem[MTILES * KT * 512 * 2 * 2];
  __bf16* ahi  = (__bf16*)smem;                 // [MTILES*KT*512]
  __bf16* alo  = ahi + MTILES * KT * 512;
  float*  zbuf = (float*)smem;                  // [MROWS*HDIM], after GEMM

  const int tid  = threadIdx.x;
  const int lane = tid & 31;
  const int wave = tid >> 5;
  const long rowbase = (long)blockIdx.x * MROWS;

  // ---- Convert x (f32, global) -> A fragments (bf16 hi/lo) in LDS ----
  // 16-bit A 16x32 layout: lane L (0..15): row m=L, K = {0..7, 16..23}
  //                        lane L+16:      row m=L, K = {8..15, 24..31}
  {
    const float4* xg = (const float4*)(x + rowbase * HDIM);
    #pragma unroll
    for (int c = 0; c < (MROWS * HDIM) / (4 * THREADS); ++c) {   // 24 chunks
      int idx4 = c * THREADS + tid;
      float4 v = xg[idx4];
      int e  = idx4 * 4;
      int m  = e / HDIM;
      int k  = e - m * HDIM;
      int mt = m >> 4;
      int ml = m & 15;
      int kt = k >> 5;
      int ko = k & 31;
      int halfsel = (ko >> 3) & 1;
      int j0 = (ko & 7) + ((ko >> 4) << 3);
      int base = ((mt * KT + kt) << 9) + (ml + halfsel * 16) * 16 + j0;
      float f[4] = {v.x, v.y, v.z, v.w};
      #pragma unroll
      for (int q = 0; q < 4; ++q) {
        __bf16 hi = (__bf16)f[q];
        ahi[base + q] = hi;
        alo[base + q] = (__bf16)(f[q] - (float)hi);
      }
    }
  }
  __syncthreads();

  // ---- GEMM: split-bf16 (hi*hi + hi*lo + lo*hi), f32 accumulate ----
  v8f acc[MTILES * NTW];
  {
    v8f vz = {0.f, 0.f, 0.f, 0.f, 0.f, 0.f, 0.f, 0.f};
    #pragma unroll
    for (int t = 0; t < MTILES * NTW; ++t) acc[t] = vz;
  }
  const int nt0 = wave * NTW;
  for (int kt = 0; kt < KT; ++kt) {
    v16bf a_hi0 = *(const v16bf*)&ahi[(0 * KT + kt) * 512 + lane * 16];
    v16bf a_lo0 = *(const v16bf*)&alo[(0 * KT + kt) * 512 + lane * 16];
    v16bf a_hi1 = *(const v16bf*)&ahi[(1 * KT + kt) * 512 + lane * 16];
    v16bf a_lo1 = *(const v16bf*)&alo[(1 * KT + kt) * 512 + lane * 16];
    #pragma unroll
    for (int t = 0; t < NTW; ++t) {
      int tile = (nt0 + t) * KT + kt;
      if (kt + 1 < KT)  // WGP-scope prefetch of next K-step of this N-tile
        prefetch_wgp((const void*)(whi + (tile + 1) * 512 + lane * 16));
      v16bf b_hi = *(const v16bf*)(whi + tile * 512 + lane * 16);
      v16bf b_lo = *(const v16bf*)(wlo + tile * 512 + lane * 16);
      // M-tile 0
      acc[t] = __builtin_amdgcn_wmma_f32_16x16x32_bf16(
          false, a_hi0, false, b_hi, (short)0, acc[t], false, false);
      acc[t] = __builtin_amdgcn_wmma_f32_16x16x32_bf16(
          false, a_hi0, false, b_lo, (short)0, acc[t], false, false);
      acc[t] = __builtin_amdgcn_wmma_f32_16x16x32_bf16(
          false, a_lo0, false, b_hi, (short)0, acc[t], false, false);
      // M-tile 1 (same B fragments -> 2x reuse of L2 bytes)
      acc[NTW + t] = __builtin_amdgcn_wmma_f32_16x16x32_bf16(
          false, a_hi1, false, b_hi, (short)0, acc[NTW + t], false, false);
      acc[NTW + t] = __builtin_amdgcn_wmma_f32_16x16x32_bf16(
          false, a_hi1, false, b_lo, (short)0, acc[NTW + t], false, false);
      acc[NTW + t] = __builtin_amdgcn_wmma_f32_16x16x32_bf16(
          false, a_lo1, false, b_hi, (short)0, acc[NTW + t], false, false);
    }
  }
  __syncthreads();   // all waves done reading A fragments; smem becomes zbuf

  // ---- Stage x (f32) -> zbuf: first half via async global->LDS (ASYNCcnt),
  // ---- second half via one TDM descriptor (TENSORcnt) issued by wave 0. ----
  {
    const char* gsrc = (const char*)(x + rowbase * HDIM);
    char* lbase = (char*)zbuf;
    // rows 0..15 (48 KB), per-lane async DMA
    #pragma unroll
    for (int i = 0; i < 12; ++i) {
      uint32_t off = (uint32_t)((i * THREADS + tid) * 16);
      uint32_t lds = (uint32_t)(uintptr_t)(lbase + off);
      const void* g = (const void*)(gsrc + off);
      asm volatile("global_load_async_to_lds_b128 %0, %1, off"
                   :: "v"(lds), "v"(g) : "memory");
    }
    asm volatile("s_wait_asynccnt 0x0" ::: "memory");
    // rows 16..31 (48 KB): one Tensor Data Mover op, 1-D tile of 12288 f32
    if (wave == 0) {
      const unsigned int nelem = 16 * HDIM;               // 12288 elements
      uint64_t gaddr = (uint64_t)(uintptr_t)(gsrc + 16 * HDIM * 4);
      uint32_t laddr = (uint32_t)(uintptr_t)(lbase + 16 * HDIM * 4);
      v4u g0;
      g0[0] = 1u;                                         // count=1, user D#
      g0[1] = laddr;                                      // lds_addr
      g0[2] = (uint32_t)(gaddr & 0xffffffffu);            // global_addr[31:0]
      g0[3] = (uint32_t)((gaddr >> 32) & 0x01ffffffu)     // global_addr[56:32]
              | (2u << 30);                               // type = 2 (image)
      v8i g1;
      g1[0] = (int)(2u << 16);          // wg_mask=0, data_size=2 (4 bytes)
      g1[1] = (int)(nelem << 16);       // tensor_dim0[15:0] in [31:16]
      g1[2] = (int)(1u << 16);          // tensor_dim0 hi=0; tensor_dim1=1
      g1[3] = (int)(nelem << 16);       // tile_dim0 in [31:16]
      g1[4] = 0;                        // tile_dim1=0 (unused), tile_dim2=0
      g1[5] = (int)nelem;               // tensor_dim0_stride[31:0]
      g1[6] = 0;
      g1[7] = 0;
      v4i zero4 = {0, 0, 0, 0};
#if __clang_major__ >= 23
      v8i zero8 = {0, 0, 0, 0, 0, 0, 0, 0};
      __builtin_amdgcn_tensor_load_to_lds(g0, g1, zero4, zero4, zero8, 0);
#else
      __builtin_amdgcn_tensor_load_to_lds(g0, g1, zero4, zero4, 0);
#endif
      __builtin_amdgcn_s_wait_tensorcnt(0);
    }
  }
  __syncthreads();

  // ---- z = x + acc + bias (LDS read-modify-write) ----
  // C/D f32 16x16 layout: VGPR r, lane L -> row m = r + (L>>4)*8, col = L&15
  {
    int mhalf = (lane >> 4) * 8;
    #pragma unroll
    for (int mt = 0; mt < MTILES; ++mt) {
      #pragma unroll
      for (int t = 0; t < NTW; ++t) {
        int n = (nt0 + t) * 16 + (lane & 15);
        float bn = bias[n];
        #pragma unroll
        for (int r = 0; r < 8; ++r) {
          int m = mt * 16 + mhalf + r;
          zbuf[m * HDIM + n] += acc[mt * NTW + t][r] + bn;
        }
      }
    }
  }
  __syncthreads();

  // ---- LayerNorm (4 rows per wave) + masked select + store ----
  #pragma unroll
  for (int rr = 0; rr < MROWS / 8; ++rr) {
    int m = wave * (MROWS / 8) + rr;
    long grow = rowbase + m;
    float s1 = 0.f, s2 = 0.f;
    #pragma unroll
    for (int i = 0; i < HDIM / 32; ++i) {
      float v = zbuf[m * HDIM + lane + i * 32];
      s1 += v; s2 += v * v;
    }
    #pragma unroll
    for (int d = 16; d > 0; d >>= 1) {       // wave32 reduction
      s1 += __shfl_xor(s1, d, 32);
      s2 += __shfl_xor(s2, d, 32);
    }
    float mean = s1 * (1.0f / HDIM);
    float var  = s2 * (1.0f / HDIM) - mean * mean;
    float rstd = rsqrtf(var + EPS);
    int msel = mask[grow];
    float* orow = out + grow * HDIM;
    const float* xrow = x + grow * HDIM;     // exact passthrough source
    #pragma unroll
    for (int i = 0; i < HDIM / 32; ++i) {
      int n = lane + i * 32;
      float zv = zbuf[m * HDIM + n];
      float y  = (zv - mean) * rstd * gamma[n] + beta[n];
      orow[n] = (msel == 1) ? y : xrow[n];
    }
  }
}

// ---------------------------------------------------------------------------
extern "C" void kernel_launch(void* const* d_in, const int* in_sizes, int n_in,
                              void* d_out, int out_size, void* d_ws, size_t ws_size,
                              hipStream_t stream) {
  const float* x     = (const float*)d_in[0];   // [64,2048,768] f32
  const int*   pmask = (const int*)  d_in[1];   // [64,2048] i32
  const float* W     = (const float*)d_in[2];   // [768,768] f32
  const float* bias  = (const float*)d_in[3];   // [768]
  const float* gamma = (const float*)d_in[4];   // [768]
  const float* beta  = (const float*)d_in[5];   // [768]
  float* out = (float*)d_out;

  __bf16* whi = (__bf16*)d_ws;
  __bf16* wlo = (__bf16*)((char*)d_ws + (size_t)HDIM * HDIM * sizeof(__bf16));

  prep_w_kernel<<<(HDIM * HDIM) / THREADS, THREADS, 0, stream>>>(W, whi, wlo);

  const int rows = 64 * 2048;
  fused_gemm_ln_kernel<<<rows / MROWS, THREADS, 0, stream>>>(
      x, pmask, whi, wlo, bias, gamma, beta, out);
}
